// LocalModel_83270825934937
// MI455X (gfx1250) — matrix operand order repaired and checked by reading, not compiled, not run on hardware
//
#include <hip/hip_runtime.h>

typedef __bf16 bf16_t;
typedef __attribute__((ext_vector_type(16))) __bf16 v16bf;
typedef __attribute__((ext_vector_type(8)))  float  v8f;

#define BB 32
#define NN 64
#define DD 256

// ---------------------------------------------------------------------------
// Kernel 1: gather embedding rows and compute pre_u / pre_i projections (f32).
// grid (B*N, 2), block 256.  y==0 -> u path, y==1 -> i path.
// pre[b,n,h] = sum_d emb[idx[b,n], d] * W1[h*512 + d + (y ? 256 : 0)]
// ---------------------------------------------------------------------------
__global__ void gather_pre_kernel(const int* __restrict__ uidx,
                                  const int* __restrict__ iidx,
                                  const float* __restrict__ emb,
                                  const float* __restrict__ W1,
                                  float* __restrict__ preU,
                                  float* __restrict__ preI) {
    __shared__ float row[DD];
    const int bn    = blockIdx.x;          // 0..2047
    const int which = blockIdx.y;          // 0=u, 1=i
    const int t     = threadIdx.x;         // 0..255 == h
    const int* idxp = which ? iidx : uidx;
    float*    pre   = which ? preI : preU;

    const int vidx = idxp[bn];
    row[t] = emb[(size_t)vidx * DD + t];
    __syncthreads();

    const float* w = W1 + (size_t)t * (2 * DD) + (which ? DD : 0);
    float acc = 0.f;
#pragma unroll 8
    for (int d = 0; d < DD; ++d) acc += row[d] * w[d];
    pre[(size_t)bn * DD + t] = acc;
}

// ---------------------------------------------------------------------------
// Kernel 2: pre-pack W2 into bf16 WMMA A-operand fragments + zero max buffers.
// Layout: w2f[((gt*8 + ks)*32 + lane)*16 + e], contiguous k order (same as B in kernel 3):
//   lane: m = lane&15, half = lane>>4 ; element e: j=e/2,
//   K = half*16 + e ; h = ks*32 + K ; g = gt*16 + m.
// grid 16, block 256 (4096 threads total).
// ---------------------------------------------------------------------------
__global__ void w2pack_kernel(const float* __restrict__ W2,
                              bf16_t* __restrict__ w2f,
                              int* __restrict__ uW, int* __restrict__ iW) {
    const int t = blockIdx.x * 256 + threadIdx.x;    // 0..4095
    if (t < BB * NN) { uW[t] = 0; iW[t] = 0; }       // scores>0, so 0 is -inf

    const int lane = t & 31;
    const int ks   = (t >> 5) & 7;
    const int gt   = t >> 8;                         // 0..15
    const int m    = lane & 15, half = lane >> 4;
    const int g    = gt * 16 + m;
    bf16_t* dst = w2f + (size_t)t * 16;
#pragma unroll
    for (int e = 0; e < 16; ++e) {
        const int j = e >> 1;
        const int K = half * 16 + e;
        const int h = ks * 32 + K;
        dst[e] = (bf16_t)W2[g * DD + h];
    }
}

// ---------------------------------------------------------------------------
// Kernel 3: the heavy one. One block per (b, a-tile16, c-tile16).
// 8 waves; wave w owns column-tiles ct = w and w+8 (ct == local a index ai,
// column ci == lane&15).  For each ct: 8 k-steps x 16 g-tiles of
// v_wmma_f32_16x16x32_bf16, B operand built from relu(preU[ai]+preI[ci]).
// Then score = sigmoid(sum_g W3[g]*relu(h2[g])), atomic-max into u/i weights.
// ---------------------------------------------------------------------------
__global__ __launch_bounds__(256, 1) void score_kernel(
        const float* __restrict__ preU, const float* __restrict__ preI,
        const bf16_t* __restrict__ w2f, const float* __restrict__ W3,
        int* __restrict__ uW, int* __restrict__ iW) {
    __shared__ float su[16 * DD];   // pre_u tile: 16 a-rows x 256 h
    __shared__ float si[16 * DD];   // pre_i tile: 16 c-rows x 256 h
    __shared__ float sw3[DD];

    const int b  = blockIdx.x;
    const int at = blockIdx.y;      // a tile (0..3)
    const int ctl = blockIdx.z;     // c tile (0..3)
    const int tid = threadIdx.x;

    const float* pu = preU + ((size_t)b * NN + at * 16) * DD;
    const float* pi = preI + ((size_t)b * NN + ctl * 16) * DD;
    for (int i = tid; i < 16 * DD; i += 256) { su[i] = pu[i]; si[i] = pi[i]; }
    if (tid < DD) sw3[tid] = W3[tid];
    __syncthreads();

    const int wave = tid >> 5;
    const int lane = tid & 31;
    const int half = lane >> 4;     // B/D lane half
    const int ci   = lane & 15;     // column == local c index

    const v16bf* afp = (const v16bf*)w2f;

    for (int rep = 0; rep < 2; ++rep) {
        const int ai = wave + rep * 8;          // column-tile == local a index
        v8f zero = {};
        v8f acc[16];
#pragma unroll
        for (int gt = 0; gt < 16; ++gt) acc[gt] = zero;

        for (int ks = 0; ks < 8; ++ks) {
            // Build B fragment: dense 32x16 bf16, K = half*16 + e, h = ks*32+K
            v16bf bfrag;
            const float* ua = &su[ai * DD + ks * 32 + half * 16];
            const float* ia = &si[ci * DD + ks * 32 + half * 16];
#pragma unroll
            for (int e = 0; e < 16; ++e) {
                float v = ua[e] + ia[e];
                v = v > 0.f ? v : 0.f;          // relu(h1)
                bfrag[e] = (bf16_t)v;
            }
#pragma unroll
            for (int gt = 0; gt < 16; ++gt) {
                v16bf afrag = afp[(size_t)((gt * 8 + ks) * 32 + lane)];
                acc[gt] = __builtin_amdgcn_wmma_f32_16x16x32_bf16(
                    /*neg_a=*/false, afrag, /*neg_b=*/false, bfrag,
                    /*c_mod=*/(short)0, acc[gt],
                    /*reuse_a=*/false, /*reuse_b=*/false);
            }
        }

        // D layout: VGPR r holds g = gt*16 + half*8 + r for column ci.
        float partial = 0.f;
#pragma unroll
        for (int gt = 0; gt < 16; ++gt) {
#pragma unroll
            for (int r = 0; r < 8; ++r) {
                float h2 = acc[gt][r];
                h2 = h2 > 0.f ? h2 : 0.f;       // relu(h2)
                partial += sw3[gt * 16 + half * 8 + r] * h2;
            }
        }
        const float total = partial + __shfl_xor(partial, 16, 32);
        const float score = 1.f / (1.f + __expf(-total));

        if (lane < 16) {
            const int a = at * 16 + ai;
            const int c = ctl * 16 + ci;
            // score > 0 => IEEE bits are monotonic => int max == float max
            atomicMax(&uW[b * NN + a], __float_as_int(score));
            atomicMax(&iW[b * NN + c], __float_as_int(score));
        }
    }
}

// ---------------------------------------------------------------------------
// Kernel 4: weighted mean features. grid (B, 2), block 256 (=d).
// out[u] at [0, B*D), out[v] at [B*D, 2*B*D).
// ---------------------------------------------------------------------------
__global__ void feat_kernel(const int* __restrict__ uidx,
                            const int* __restrict__ iidx,
                            const float* __restrict__ emb,
                            const int* __restrict__ uW,
                            const int* __restrict__ iW,
                            float* __restrict__ out) {
    const int b = blockIdx.x, which = blockIdx.y, d = threadIdx.x;
    const int* idxp = which ? iidx : uidx;
    const int* W    = which ? iW   : uW;
    float s = 0.f;
#pragma unroll 4
    for (int n = 0; n < NN; ++n) {
        const int vidx = idxp[b * NN + n];
        s += emb[(size_t)vidx * DD + d] * __int_as_float(W[b * NN + n]);
    }
    out[(size_t)which * BB * DD + b * DD + d] = s * (1.f / (float)NN);
}

// ---------------------------------------------------------------------------
extern "C" void kernel_launch(void* const* d_in, const int* in_sizes, int n_in,
                              void* d_out, int out_size, void* d_ws, size_t ws_size,
                              hipStream_t stream) {
    // input order: params, u_matrix, i_matrix, emb, W1, W2, W3
    const int*   uidx = (const int*)d_in[1];
    const int*   iidx = (const int*)d_in[2];
    const float* emb  = (const float*)d_in[3];
    const float* W1   = (const float*)d_in[4];
    const float* W2   = (const float*)d_in[5];
    const float* W3   = (const float*)d_in[6];
    float* out = (float*)d_out;

    // workspace layout
    const size_t rows = (size_t)BB * NN;            // 2048
    float*  preU = (float*)d_ws;                    // rows*DD f32
    float*  preI = preU + rows * DD;
    bf16_t* w2f  = (bf16_t*)(preI + rows * DD);     // DD*DD bf16 fragments
    int*    uW   = (int*)(w2f + DD * DD);           // rows ints
    int*    iW   = uW + rows;

    gather_pre_kernel<<<dim3(BB * NN, 2), 256, 0, stream>>>(
        uidx, iidx, emb, W1, preU, preI);
    w2pack_kernel<<<dim3(16), 256, 0, stream>>>(W2, w2f, uW, iW);
    score_kernel<<<dim3(BB, 4, 4), 256, 0, stream>>>(
        preU, preI, w2f, W3, uW, iW);
    feat_kernel<<<dim3(BB, 2), 256, 0, stream>>>(
        uidx, iidx, emb, uW, iW, out);
}
